// VideoMAESelfAttention_46308337385912
// MI455X (gfx1250) — matrix operand order repaired and checked by reading, not compiled, not run on hardware
//
#include <hip/hip_runtime.h>
#include <hip/hip_bf16.h>
#include <math.h>

typedef __attribute__((ext_vector_type(16))) __bf16 v16bf;
typedef __attribute__((ext_vector_type(8)))  __bf16 v8bf;
typedef __attribute__((ext_vector_type(8)))  float  v8f;

// Problem constants (VideoMAE base)
constexpr int Bn = 4;            // batch
constexpr int Sn = 1568;         // sequence
constexpr int Dn = 768;          // model dim
constexpr int Hn = 12;           // heads
constexpr int Dh = 64;           // head dim

static __device__ __forceinline__ v8f wmma_bf16(v16bf a, v16bf b, v8f c) {
  // D = A(16x32 bf16) x B(32x16 bf16) + C(16x16 f32)
  return __builtin_amdgcn_wmma_f32_16x16x32_bf16(false, a, false, b, (short)0, c,
                                                 false, false);
}

// A-operand (16x32, MxK) per ISA 7.12.2: lane (lo,hi) holds row M=lo,
// elements e=0..7 -> K = k0 + 8*hi + e ; e=8..15 -> K = k0 + 16 + 8*hi + (e-8)
static __device__ __forceinline__ v16bf load_a16x32(const __bf16* row, int k0, int hi) {
  v8bf l = *reinterpret_cast<const v8bf*>(row + k0 + 8 * hi);
  v8bf h = *reinterpret_cast<const v8bf*>(row + k0 + 16 + 8 * hi);
  return __builtin_shufflevector(l, h, 0, 1, 2, 3, 4, 5, 6, 7,
                                 8, 9, 10, 11, 12, 13, 14, 15);
}

// B-operand (32x16, KxN): lane (lo,hi) holds column N=lo,
// elements e=0..7 -> K = k0 + 8*hi + e ; e=8..15 -> K = k0 + 16 + 8*hi + (e-8)
static __device__ __forceinline__ v16bf load_b32x16(const __bf16* row, int k0, int hi) {
  v8bf l = *reinterpret_cast<const v8bf*>(row + k0 + 8 * hi);
  v8bf h = *reinterpret_cast<const v8bf*>(row + k0 + 16 + 8 * hi);
  return __builtin_shufflevector(l, h, 0, 1, 2, 3, 4, 5, 6, 7,
                                 8, 9, 10, 11, 12, 13, 14, 15);
}

// ---------------------------------------------------------------------------
// fp32 -> bf16 elementwise conversion (n divisible by 4)
// ---------------------------------------------------------------------------
__global__ void cvt_f32_to_bf16(const float* __restrict__ src,
                                __bf16* __restrict__ dst, int n) {
  int i = 4 * (blockIdx.x * blockDim.x + threadIdx.x);
  if (i < n) {
    float4 v = *reinterpret_cast<const float4*>(src + i);
    dst[i + 0] = (__bf16)v.x;
    dst[i + 1] = (__bf16)v.y;
    dst[i + 2] = (__bf16)v.z;
    dst[i + 3] = (__bf16)v.w;
  }
}

// ---------------------------------------------------------------------------
// Fused QKV projection: out = X @ W^T (+bias), written as
//   Q  [B,H,S,Dh] bf16 (scaled by 1/sqrt(Dh))
//   K  [B,H,S,Dh] bf16
//   Vt [B,H,Dh,S] bf16 (transposed so PV GEMM's B operand is contiguous)
// One wave per 16x64 output tile; 24 K-steps of 32, 4 WMMAs each.
// ---------------------------------------------------------------------------
__global__ __launch_bounds__(256) void qkv_proj(const __bf16* __restrict__ X,
                                                const __bf16* __restrict__ W3,
                                                const float* __restrict__ qb,
                                                const float* __restrict__ vb,
                                                __bf16* __restrict__ Qo,
                                                __bf16* __restrict__ Ko,
                                                __bf16* __restrict__ Vto) {
  const int lane = threadIdx.x & 31;
  const int lo = lane & 15;
  const int hi = lane >> 4;
  const int wave = (blockIdx.x * blockDim.x + threadIdx.x) >> 5;

  constexpr int RT = (Bn * Sn) / 16;  // 392 row tiles
  constexpr int CB = Dn / 64;         // 12 column blocks (== head index)
  const int mat = wave / (RT * CB);   // 0=Q, 1=K, 2=V
  int rem = wave % (RT * CB);
  const int cb = rem / RT;
  const int rt = rem % RT;

  const __bf16* Wm = W3 + (size_t)mat * Dn * Dn;
  const __bf16* xrow = X + (size_t)(rt * 16 + lo) * Dn;  // A row M=lo

  v8f acc[4] = {};
  for (int k0 = 0; k0 < Dn; k0 += 32) {
    v16bf a = load_a16x32(xrow, k0, hi);
#pragma unroll
    for (int t = 0; t < 4; ++t) {
      // out[m][n] = sum_k X[m][k] * W[n][k]; B[k][n] = W[n][k] -> W row n
      const __bf16* wrow = Wm + (size_t)(cb * 64 + 16 * t + lo) * Dn;
      v16bf bmat = load_b32x16(wrow, k0, hi);
      acc[t] = wmma_bf16(a, bmat, acc[t]);
    }
  }

  // C-layout: acc[t][i] = (row rt*16 + i + 8*hi, col cb*64 + 16*t + lo)
  const int bb = (rt * 16) / Sn;       // 1568 % 16 == 0: whole tile same batch
  const int sbase = rt * 16 - bb * Sn;
#pragma unroll
  for (int t = 0; t < 4; ++t) {
    const int n = cb * 64 + 16 * t + lo;
    const int d = 16 * t + lo;
    const float bias = (mat == 0) ? qb[n] : (mat == 2 ? vb[n] : 0.f);
#pragma unroll
    for (int i = 0; i < 8; ++i) {
      const int s = sbase + i + 8 * hi;
      float v = acc[t][i] + bias;
      if (mat == 0) v *= 0.125f;  // fold 1/sqrt(64) into Q
      __bf16 bv = (__bf16)v;
      if (mat == 0)
        Qo[(((size_t)(bb * Hn + cb)) * Sn + s) * Dh + d] = bv;
      else if (mat == 1)
        Ko[(((size_t)(bb * Hn + cb)) * Sn + s) * Dh + d] = bv;
      else
        Vto[(((size_t)(bb * Hn + cb)) * Dh + d) * Sn + s] = bv;
    }
  }
}

// ---------------------------------------------------------------------------
// Flash attention: one wave per (b, h, 32-query tile).
// Two 16-query groups share the K/Vt operand registers (halves VMEM/WMMA and
// doubles independent WMMA chains for co-execution with softmax VALU).
// Scores computed TRANSPOSED (S^T = K * Q^T) so the probability tile lands
// in-register exactly in the A-operand layout for the PV GEMM (no LDS).
// Online softmax state (m, l) is a per-lane scalar (query = lo).
// ---------------------------------------------------------------------------
__global__ __launch_bounds__(256) void flash_attn(const __bf16* __restrict__ Qm,
                                                  const __bf16* __restrict__ Km,
                                                  const __bf16* __restrict__ Vtm,
                                                  float* __restrict__ out) {
  const int lane = threadIdx.x & 31;
  const int lo = lane & 15;
  const int hi = lane >> 4;
  const int wave = (blockIdx.x * blockDim.x + threadIdx.x) >> 5;

  constexpr int QT = Sn / 32;  // 49 query tiles of 32
  const int b = wave / (Hn * QT);
  int rem = wave % (Hn * QT);
  const int h = rem / QT;
  const int qbase = (rem % QT) * 32;

  const __bf16* Qh = Qm + ((size_t)(b * Hn + h)) * Sn * Dh;
  const __bf16* Kh = Km + ((size_t)(b * Hn + h)) * Sn * Dh;
  const __bf16* Vh = Vtm + ((size_t)(b * Hn + h)) * Dh * Sn;

  // B operand of S^T per query group: B[d][n] = Q[n][d] -> lane holds Q row.
  v16bf bq0[2], bq1[2];
#pragma unroll
  for (int g = 0; g < 2; ++g) {
    const __bf16* qrow = Qh + (size_t)(qbase + 16 * g + lo) * Dh;
    bq0[g] = load_b32x16(qrow, 0, hi);
    bq1[g] = load_b32x16(qrow, 32, hi);
  }

  float m[2] = {-__builtin_inff(), -__builtin_inff()};
  float lsum[2] = {0.f, 0.f};
  v8f acc[2][4] = {};  // O per group: C-layout rows = queries, cols = d
  v16bf pa[2];         // probability tiles (A-operand layout)

  for (int kc = 0; kc < Sn; kc += 32) {
    // A operand: K rows (keys), shared by both query groups.
    const __bf16* k0r = Kh + (size_t)(kc + lo) * Dh;
    const __bf16* k1r = Kh + (size_t)(kc + 16 + lo) * Dh;
    v16bf a0 = load_a16x32(k0r, 0, hi);
    v16bf a1 = load_a16x32(k0r, 32, hi);
    v16bf a2 = load_a16x32(k1r, 0, hi);
    v16bf a3 = load_a16x32(k1r, 32, hi);

#pragma unroll
    for (int g = 0; g < 2; ++g) {
      v8f t1 = {}, t2 = {};
      t1 = wmma_bf16(a0, bq0[g], t1);
      t1 = wmma_bf16(a1, bq1[g], t1);
      t2 = wmma_bf16(a2, bq0[g], t2);
      t2 = wmma_bf16(a3, bq1[g], t2);
      // t1[i]: (key = i + 8*hi,      query = 16*g + lo)
      // t2[i]: (key = 16 + i + 8*hi, query = 16*g + lo)

      float cm = fmaxf(t1[0], t2[0]);
#pragma unroll
      for (int i = 1; i < 8; ++i) cm = fmaxf(cm, fmaxf(t1[i], t2[i]));
      cm = fmaxf(cm, __shfl_xor(cm, 16, 32));  // merge hi-halves of key range
      const float mn = fmaxf(m[g], cm);
      const float es = __expf(m[g] - mn);
      m[g] = mn;

      // P directly in A-operand layout
      float ps = 0.f;
#pragma unroll
      for (int i = 0; i < 8; ++i) {
        float p0 = __expf(t1[i] - mn);
        float p1 = __expf(t2[i] - mn);
        ps += p0 + p1;
        pa[g][i] = (__bf16)p0;
        pa[g][i + 8] = (__bf16)p1;
      }
      ps += __shfl_xor(ps, 16, 32);
      lsum[g] = lsum[g] * es + ps;

      // Rescale O: acc row i holds query i + 8*hi -> broadcast es from lane
#pragma unroll
      for (int i = 0; i < 8; ++i) {
        float esr = __shfl(es, i + 8 * hi, 32);
#pragma unroll
        for (int t = 0; t < 4; ++t) acc[g][t][i] *= esr;
      }
    }

    // O += P(16x32) x V(32x64); Vt tile shared by both groups.
#pragma unroll
    for (int t = 0; t < 4; ++t) {
      v16bf bv = load_a16x32(Vh + (size_t)(16 * t + lo) * Sn, kc, hi);
#pragma unroll
      for (int g = 0; g < 2; ++g) acc[g][t] = wmma_bf16(pa[g], bv, acc[g][t]);
    }
  }

#pragma unroll
  for (int g = 0; g < 2; ++g) {
    const float rl = 1.f / lsum[g];
#pragma unroll
    for (int i = 0; i < 8; ++i) {
      float rlr = __shfl(rl, i + 8 * hi, 32);
      const int s = qbase + 16 * g + i + 8 * hi;
      float* op = out + ((size_t)b * Sn + s) * Dn + h * Dh;
#pragma unroll
      for (int t = 0; t < 4; ++t) op[16 * t + lo] = acc[g][t][i] * rlr;
    }
  }
}

// ---------------------------------------------------------------------------
extern "C" void kernel_launch(void* const* d_in, const int* in_sizes, int n_in,
                              void* d_out, int out_size, void* d_ws, size_t ws_size,
                              hipStream_t stream) {
  (void)in_sizes; (void)n_in; (void)out_size; (void)ws_size;
  const float* hs = (const float*)d_in[0];
  const float* Wq = (const float*)d_in[1];
  const float* Wk = (const float*)d_in[2];
  const float* Wv = (const float*)d_in[3];
  const float* qb = (const float*)d_in[4];
  const float* vb = (const float*)d_in[5];
  float* out = (float*)d_out;

  const size_t nX = (size_t)Bn * Sn * Dn;  // 4,816,896 elems
  const size_t nW = (size_t)Dn * Dn;       // 589,824 elems

  // Workspace layout (bf16): X | Wq,Wk,Wv | Q | K | Vt  (~40.1 MB total)
  char* ws = (char*)d_ws;
  __bf16* Xbf = reinterpret_cast<__bf16*>(ws);
  __bf16* W3  = reinterpret_cast<__bf16*>(ws + 2 * nX);
  __bf16* Qbf = reinterpret_cast<__bf16*>(ws + 2 * (nX + 3 * nW));
  __bf16* Kbf = reinterpret_cast<__bf16*>(ws + 2 * (2 * nX + 3 * nW));
  __bf16* Vtb = reinterpret_cast<__bf16*>(ws + 2 * (3 * nX + 3 * nW));

  const int TPB = 256;
  cvt_f32_to_bf16<<<(int)((nX / 4 + TPB - 1) / TPB), TPB, 0, stream>>>(hs, Xbf, (int)nX);
  cvt_f32_to_bf16<<<(int)((nW / 4 + TPB - 1) / TPB), TPB, 0, stream>>>(Wq, W3, (int)nW);
  cvt_f32_to_bf16<<<(int)((nW / 4 + TPB - 1) / TPB), TPB, 0, stream>>>(Wk, W3 + nW, (int)nW);
  cvt_f32_to_bf16<<<(int)((nW / 4 + TPB - 1) / TPB), TPB, 0, stream>>>(Wv, W3 + 2 * nW, (int)nW);

  // 3 matrices x 12 col-blocks x 392 row-tiles = 14112 waves -> 1764 blocks
  qkv_proj<<<1764, 256, 0, stream>>>(Xbf, W3, qb, vb, Qbf, Kbf, Vtb);

  // 4 x 12 x 49 = 2352 waves -> 294 blocks of 8 waves
  flash_attn<<<294, 256, 0, stream>>>(Qbf, Kbf, Vtb, out);
}
